// UNet_41515153883163
// MI455X (gfx1250) — matrix operand order repaired and checked by reading, not compiled, not run on hardware
//
#include <hip/hip_runtime.h>
#include <hip/hip_bf16.h>
#include <stdint.h>

#define CEILDIV(a,b) (((a)+(b)-1)/(b))

typedef __attribute__((ext_vector_type(16))) __bf16 v16bf;
typedef __attribute__((ext_vector_type(8)))  __bf16 v8bf;
typedef __attribute__((ext_vector_type(8)))  float  v8f;

static const int PLh[5]  = {32, 64, 128, 256, 512};
static const int NSh[5]  = {131072, 32768, 8192, 2048, 512};
#define N0 131072
#define KNN 16
#define KC  128   // K-chunk staged in LDS (bf16 elements)
#define PADU 8    // per-row LDS pad (ushorts, 16B) to spread banks

__device__ __forceinline__ unsigned short f2bf(float f) {
  union { float f; unsigned int u; } v; v.f = f;
  unsigned int u = v.u;
  unsigned int r = (u + 0x7FFFu + ((u >> 16) & 1u)) >> 16;
  return (unsigned short)r;
}

// -------- generic f32 -> bf16 row convert (dcols may pad; pad pre-zeroed) ----
__global__ void k_cvt_bf16(const float* __restrict__ src, unsigned short* __restrict__ dst,
                           int rows, int scols, int dcols) {
  int idx = blockIdx.x * blockDim.x + threadIdx.x;
  if (idx >= rows * scols) return;
  int r = idx / scols, c = idx % scols;
  dst[(size_t)r * dcols + c] = f2bf(src[idx]);
}

// -------- pack weight W[Fin][Fout] (f32) into Wt[Fout][KdimP] (bf16) at koff --
__global__ void k_pack_wt(const float* __restrict__ W, unsigned short* __restrict__ Wt,
                          int Fin, int Fout, int KdimP, int koff) {
  int idx = blockIdx.x * blockDim.x + threadIdx.x;
  if (idx >= Fin * Fout) return;
  int k = idx / Fout, n = idx % Fout;
  Wt[(size_t)n * KdimP + koff + k] = f2bf(W[(size_t)k * Fout + n]);
}

// -------- fused gather + mean: abuf[i] = bf16([ h[i] | mean_k h[src[i,k]] ]) --
__global__ void k_gather_pack(const float* __restrict__ h, const int* __restrict__ src,
                              unsigned short* __restrict__ abuf, int n, int F) {
  int Fv = F >> 2;
  int idx = blockIdx.x * blockDim.x + threadIdx.x;
  if (idx >= n * Fv) return;
  int node = idx / Fv, f4 = idx % Fv;
  const float4* hv4 = (const float4*)h;
  float4 self = hv4[(size_t)node * Fv + f4];
  float4 m = make_float4(0.f, 0.f, 0.f, 0.f);
  const int* s = src + (size_t)node * KNN;
#pragma unroll
  for (int k = 0; k < KNN; ++k) {
    float4 v = hv4[(size_t)s[k] * Fv + f4];
    m.x += v.x; m.y += v.y; m.z += v.z; m.w += v.w;
  }
  const float inv = 1.0f / (float)KNN;
  unsigned short* o = abuf + (size_t)node * (2 * F);
  int c = f4 * 4;
  o[c + 0] = f2bf(self.x); o[c + 1] = f2bf(self.y);
  o[c + 2] = f2bf(self.z); o[c + 3] = f2bf(self.w);
  o[F + c + 0] = f2bf(m.x * inv); o[F + c + 1] = f2bf(m.y * inv);
  o[F + c + 2] = f2bf(m.z * inv); o[F + c + 3] = f2bf(m.w * inv);
}

// -------- WMMA GEMM: Y[n x Fout] = A[n x Kdim](bf16) * Wt[Fout x Kdim]^T -----
// Block = 8 waves (256 thr) x 8 row-tiles sharing an NCT*16-column weight panel
// staged through LDS in KC-chunks. Each wave register-blocks NCT column tiles
// (compile-time constant -> accumulators stay in fixed VGPRs, no spills),
// reusing its A fragment NCT x per K-step. Optional fused BN sum/sumsq atomics.
template <int NCT>
__global__ void k_wmma_gemm(const unsigned short* __restrict__ A,
                            const unsigned short* __restrict__ Wt,
                            float* __restrict__ Y,
                            int n, int Kdim, int Fout, int ldY,
                            float* __restrict__ stats) {
  __shared__ unsigned short bpanel[NCT * 16][KC + PADU];

  int rT = n >> 4;
  int col0g   = blockIdx.x * (NCT * 16);         // first column of this group
  int waveId  = threadIdx.x >> 5;
  int lane    = threadIdx.x & 31;
  int rowTile = blockIdx.y * 8 + waveId;
  int row0    = rowTile << 4;
  int half    = lane >> 4;                       // ISA 16-bit fragment halves
  int r       = lane & 15;
  bool rowValid = rowTile < rT;                  // wave-uniform

  const unsigned short* arow = A + (size_t)(row0 + r) * Kdim;
  v8f acc[NCT] = {};

  for (int k0 = 0; k0 < Kdim; k0 += KC) {
    int kc = min(KC, Kdim - k0);                 // multiple of 32
    // ---- cooperative stage of (NCT*16) x kc weight panel into LDS ----
    int kv = kc >> 3;                            // 8-ushort vectors per row
    for (int v = threadIdx.x; v < NCT * 16 * kv; v += blockDim.x) {
      int rowp = v / kv;
      int kp   = (v - rowp * kv) << 3;
      *(v8bf*)(void*)&bpanel[rowp][kp] =
          *(const v8bf*)(const void*)(Wt + (size_t)(col0g + rowp) * Kdim + k0 + kp);
    }
    __syncthreads();

    if (rowValid) {
      for (int kk = 0; kk < kc; kk += 32) {
        // A fragment (ISA 16x32 bf16 layout)
        v8bf alo = *(const v8bf*)(const void*)(arow + k0 + kk + half * 8);
        v8bf ahi = *(const v8bf*)(const void*)(arow + k0 + kk + 16 + half * 8);
        v16bf a = __builtin_shufflevector(alo, ahi,
                    0,1,2,3,4,5,6,7,8,9,10,11,12,13,14,15);
#pragma unroll
        for (int cc = 0; cc < NCT; ++cc) {
          const unsigned short* bp = &bpanel[cc * 16 + r][kk + half * 8];
          v8bf blo = *(const v8bf*)(const void*)bp;
          v8bf bhi = *(const v8bf*)(const void*)(bp + 16);
          v16bf b = __builtin_shufflevector(blo, bhi,
                      0,1,2,3,4,5,6,7,8,9,10,11,12,13,14,15);
          acc[cc] = __builtin_amdgcn_wmma_f32_16x16x32_bf16(
                      false, a, false, b, (short)0, acc[cc], false, false);
        }
      }
    }
    __syncthreads();
  }

  if (!rowValid) return;
  // C/D layout: VGPR j -> row (j + 8*half), column r
#pragma unroll
  for (int cc = 0; cc < NCT; ++cc) {
    int col = col0g + cc * 16 + r;
    float s = 0.f, q = 0.f;
#pragma unroll
    for (int j = 0; j < 8; ++j) {
      int m = row0 + half * 8 + j;
      float v = acc[cc][j];
      Y[(size_t)m * ldY + col] = v;
      s += v; q += v * v;
    }
    if (stats) {
      atomicAdd(&stats[col], s);
      atomicAdd(&stats[512 + col], q);
    }
  }
}

// -------- BatchNorm (+ReLU or +ELU [+residual]) --------------------------
__global__ void k_bn_act(const float* __restrict__ y, float* __restrict__ out,
                         const float* __restrict__ base,
                         const float* __restrict__ stats,
                         const float* __restrict__ gamma, const float* __restrict__ beta,
                         int n, int F, int elu_mode) {
  int idx = blockIdx.x * blockDim.x + threadIdx.x;
  if (idx >= n * F) return;
  int f = idx % F;
  float inv_n = 1.0f / (float)n;
  float mean = stats[f] * inv_n;
  float var  = stats[512 + f] * inv_n - mean * mean;
  float x = (y[idx] - mean) * rsqrtf(var + 1e-5f) * gamma[f] + beta[f];
  x = elu_mode ? (x > 0.f ? x : (__expf(x) - 1.f)) : fmaxf(x, 0.f);
  out[idx] = base ? base[idx] + x : x;
}

// -------- up-scatter + skip fuse: h = elu(scatter(up) + skipY) ------------
__global__ void k_up_fuse(const float* __restrict__ upY, const float* __restrict__ skipY,
                          float* __restrict__ h, int nCoarse, int n, int F) {
  int idx = blockIdx.x * blockDim.x + threadIdx.x;
  if (idx >= n * F) return;
  int row = idx / F;
  float u = (row < nCoarse) ? upY[idx] : 0.f;
  float x = u + skipY[idx];
  h[idx] = x > 0.f ? x : (__expf(x) - 1.f);
}

__global__ void k_add(float* __restrict__ a, const float* __restrict__ b, int count) {
  int idx = blockIdx.x * blockDim.x + threadIdx.x;
  if (idx < count) a[idx] += b[idx];
}

__global__ void k_copy13(const float* __restrict__ y16, float* __restrict__ out, int n) {
  int idx = blockIdx.x * blockDim.x + threadIdx.x;
  if (idx >= n * 13) return;
  int row = idx / 13, c = idx % 13;
  out[idx] = y16[(size_t)row * 16 + c];
}

// =========================================================================
extern "C" void kernel_launch(void* const* d_in, const int* in_sizes, int n_in,
                              void* d_out, int out_size, void* d_ws, size_t ws_size,
                              hipStream_t stream) {
  (void)in_sizes; (void)n_in; (void)out_size; (void)ws_size;
  auto F32 = [&](int i) { return (const float*)d_in[i]; };
  auto I32 = [&](int i) { return (const int*)d_in[i]; };

  // ---- flat leaf indices (jax dict-sorted order) ----
  // 0 feat | 1-8 bottleneck (2x beta,gamma,w_nbr,w_self) | 9 cls_b1 10 cls_beta
  // 11 cls_gamma 12 cls_w1 13 cls_w2 | 14-17 down_w | 18 emb_beta 19 emb_gamma
  // 20 emb_w | 21-52 mp_down (4 stages x 2 layers x {beta,gamma,w_nbr,w_self})
  // 53-84 mp_up | 85-88 skip_w | 89-92 up_w | 93-97 src0..4
  const float* feat = F32(0);
  const int* srcs[5]; for (int l = 0; l < 5; ++l) srcs[l] = I32(93 + l);

  // ---- workspace bump allocator ----
  char* wsBase = (char*)d_ws;
  size_t off = 0;
  auto alloc = [&](size_t bytes) -> void* {
    void* p = wsBase + off;
    off = (off + bytes + 255) & ~(size_t)255;
    return p;
  };

  // weight arena (memset once: covers emb K-pad and cls2 row-pad)
  size_t wStart = off;
  unsigned short* embWt  = (unsigned short*)alloc(32 * 32 * 2);
  unsigned short* cls1Wt = (unsigned short*)alloc(32 * 32 * 2);
  unsigned short* cls2Wt = (unsigned short*)alloc(16 * 32 * 2);
  unsigned short *mpDownWt[4][2], *mpUpWt[4][2], *botWt[2];
  unsigned short *downWt[4], *upWt[4], *skipWt[4];
  for (int l = 0; l < 4; ++l)
    for (int ly = 0; ly < 2; ++ly) {
      mpDownWt[l][ly] = (unsigned short*)alloc((size_t)PLh[l] * 2 * PLh[l] * 2);
      mpUpWt[l][ly]   = (unsigned short*)alloc((size_t)PLh[l] * 2 * PLh[l] * 2);
    }
  for (int ly = 0; ly < 2; ++ly) botWt[ly] = (unsigned short*)alloc((size_t)512 * 1024 * 2);
  for (int l = 0; l < 4; ++l) {
    downWt[l] = (unsigned short*)alloc((size_t)PLh[l + 1] * PLh[l] * 2);
    upWt[l]   = (unsigned short*)alloc((size_t)PLh[l] * PLh[l + 1] * 2);
    skipWt[l] = (unsigned short*)alloc((size_t)PLh[l] * PLh[l] * 2);
  }
  size_t wBytes = off - wStart;

  float* stats = (float*)alloc(20 * 1024 * 4);       // 20 BN slots x (sum[512]|sumsq[512])
  unsigned short* featb = (unsigned short*)alloc((size_t)N0 * 32 * 2);  // padded K 6->32
  float *h[5], *hs[5];
  for (int l = 0; l < 5; ++l) {
    h[l]  = (float*)alloc((size_t)NSh[l] * PLh[l] * 4);
    hs[l] = (float*)alloc((size_t)NSh[l] * PLh[l] * 4);
  }
  unsigned short* abuf = (unsigned short*)alloc((size_t)N0 * 64 * 2);   // max n*2F
  float* ybuf = (float*)alloc((size_t)N0 * 32 * 4);                     // max n*F
  unsigned short* hb = (unsigned short*)alloc((size_t)N0 * 32 * 2);     // bf16 staging
  float* tmpUp   = (float*)alloc((size_t)NSh[1] * 32 * 4);
  float* tmpSkip = (float*)alloc((size_t)N0 * 32 * 4);

  // ---- zero pads / stats ----
  hipMemsetAsync(wsBase + wStart, 0, wBytes, stream);
  hipMemsetAsync(stats, 0, 20 * 1024 * 4, stream);
  hipMemsetAsync(featb, 0, (size_t)N0 * 32 * 2, stream);

  // ---- pack weights (bf16, transposed, mp layers K-stacked [W_self;W_nbr]) ----
  auto pack = [&](const float* W, unsigned short* Wt, int Fin, int Fout, int Kp, int koff) {
    k_pack_wt<<<CEILDIV(Fin * Fout, 256), 256, 0, stream>>>(W, Wt, Fin, Fout, Kp, koff);
  };
  pack(F32(20), embWt, 6, 32, 32, 0);
  pack(F32(12), cls1Wt, 32, 32, 32, 0);
  pack(F32(13), cls2Wt, 32, 13, 32, 0);
  for (int l = 0; l < 4; ++l) {
    int F = PLh[l];
    for (int ly = 0; ly < 2; ++ly) {
      int bd = 21 + l * 8 + ly * 4;   // mp_down leaf base: beta,gamma,w_nbr,w_self
      pack(F32(bd + 3), mpDownWt[l][ly], F, F, 2 * F, 0);
      pack(F32(bd + 2), mpDownWt[l][ly], F, F, 2 * F, F);
      int bu = 53 + l * 8 + ly * 4;
      pack(F32(bu + 3), mpUpWt[l][ly], F, F, 2 * F, 0);
      pack(F32(bu + 2), mpUpWt[l][ly], F, F, 2 * F, F);
    }
    pack(F32(14 + l), downWt[l], PLh[l], PLh[l + 1], PLh[l], 0);
    pack(F32(89 + l), upWt[l], PLh[l + 1], PLh[l], PLh[l + 1], 0);
    pack(F32(85 + l), skipWt[l], PLh[l], PLh[l], PLh[l], 0);
  }
  for (int ly = 0; ly < 2; ++ly) {
    int bb = 1 + ly * 4;
    pack(F32(bb + 3), botWt[ly], 512, 512, 1024, 0);
    pack(F32(bb + 2), botWt[ly], 512, 512, 1024, 512);
  }

  // NCT is launch-uniform: cT is 1, 2 or a multiple of 4 for every GEMM here.
  auto gemm = [&](const unsigned short* A, const unsigned short* Wt, float* Y,
                  int n, int Kdim, int Fout, int ldY, float* st) {
    int cT = Fout >> 4;
    int nct = cT < 4 ? cT : 4;
    dim3 grid(CEILDIV(cT, nct), CEILDIV(n >> 4, 8));
    switch (nct) {
      case 1: k_wmma_gemm<1><<<grid, 256, 0, stream>>>(A, Wt, Y, n, Kdim, Fout, ldY, st); break;
      case 2: k_wmma_gemm<2><<<grid, 256, 0, stream>>>(A, Wt, Y, n, Kdim, Fout, ldY, st); break;
      default: k_wmma_gemm<4><<<grid, 256, 0, stream>>>(A, Wt, Y, n, Kdim, Fout, ldY, st); break;
    }
  };
  auto mp_layer = [&](float* hl, const int* src, int n, int F, unsigned short* wt,
                      const float* gamma, const float* beta, int slot) {
    k_gather_pack<<<CEILDIV(n * (F / 4), 256), 256, 0, stream>>>(hl, src, abuf, n, F);
    gemm(abuf, wt, ybuf, n, 2 * F, F, F, stats + slot * 1024);
    k_bn_act<<<CEILDIV(n * F, 256), 256, 0, stream>>>(ybuf, hl, hl, stats + slot * 1024,
                                                      gamma, beta, n, F, 1);
  };

  // ---- embedding: relu(bn(feat @ emb_w)) ----
  k_cvt_bf16<<<CEILDIV(N0 * 6, 256), 256, 0, stream>>>(feat, featb, N0, 6, 32);
  gemm(featb, embWt, ybuf, N0, 32, 32, 32, stats + 0);
  k_bn_act<<<CEILDIV(N0 * 32, 256), 256, 0, stream>>>(ybuf, h[0], nullptr, stats + 0,
                                                      F32(19), F32(18), N0, 32, 0);

  // ---- contracting path ----
  for (int l = 0; l < 4; ++l) {
    int n = NSh[l], F = PLh[l];
    for (int ly = 0; ly < 2; ++ly) {
      int bd = 21 + l * 8 + ly * 4;
      mp_layer(h[l], srcs[l], n, F, mpDownWt[l][ly], F32(bd + 1), F32(bd), 1 + l * 2 + ly);
    }
    hipMemcpyAsync(hs[l], h[l], (size_t)n * F * 4, hipMemcpyDeviceToDevice, stream);
    int n2 = NSh[l + 1];
    k_cvt_bf16<<<CEILDIV(n2 * F, 256), 256, 0, stream>>>(h[l], hb, n2, F, F);
    gemm(hb, downWt[l], h[l + 1], n2, F, PLh[l + 1], PLh[l + 1], nullptr);
  }

  // ---- bottleneck ----
  for (int ly = 0; ly < 2; ++ly) {
    int bb = 1 + ly * 4;
    mp_layer(h[4], srcs[4], 512, 512, botWt[ly], F32(bb + 1), F32(bb), 9 + ly);
  }

  // ---- expanding path ----
  for (int u = 0; u < 4; ++u) {
    int l = 3 - u, n = NSh[l], nC = NSh[l + 1], F = PLh[l], Fc = PLh[l + 1];
    k_cvt_bf16<<<CEILDIV(nC * Fc, 256), 256, 0, stream>>>(h[l + 1], hb, nC, Fc, Fc);
    gemm(hb, upWt[l], tmpUp, nC, Fc, F, F, nullptr);
    k_cvt_bf16<<<CEILDIV(n * F, 256), 256, 0, stream>>>(hs[l], hb, n, F, F);
    gemm(hb, skipWt[l], tmpSkip, n, F, F, F, nullptr);
    k_up_fuse<<<CEILDIV(n * F, 256), 256, 0, stream>>>(tmpUp, tmpSkip, h[l], nC, n, F);
    for (int ly = 0; ly < 2; ++ly) {
      int bu = 53 + l * 8 + ly * 4;
      mp_layer(h[l], srcs[l], n, F, mpUpWt[l][ly], F32(bu + 1), F32(bu), 11 + u * 2 + ly);
    }
    k_add<<<CEILDIV(n * F, 256), 256, 0, stream>>>(h[l], hs[l], n * F);
  }

  // ---- classifier (cls_b1 cancels inside BN; relu(bn(h@w1)) @ w2, Fout pad 13->16) ----
  k_cvt_bf16<<<CEILDIV(N0 * 32, 256), 256, 0, stream>>>(h[0], hb, N0, 32, 32);
  gemm(hb, cls1Wt, ybuf, N0, 32, 32, 32, stats + 19 * 1024);
  k_bn_act<<<CEILDIV(N0 * 32, 256), 256, 0, stream>>>(ybuf, hs[0], nullptr, stats + 19 * 1024,
                                                      F32(11), F32(10), N0, 32, 0);
  k_cvt_bf16<<<CEILDIV(N0 * 32, 256), 256, 0, stream>>>(hs[0], hb, N0, 32, 32);
  gemm(hb, cls2Wt, ybuf, N0, 32, 16, 16, nullptr);
  k_copy13<<<CEILDIV(N0 * 13, 256), 256, 0, stream>>>(ybuf, (float*)d_out, N0);
}
